// MultiheadAttention_7816840478952
// MI455X (gfx1250) — compile-verified, matrix operand order checked by reading, not run on hardware
//
#include <hip/hip_runtime.h>
#include <stdint.h>

#define T_LEN 1024
#define S_LEN 1024
#define BSZ   4
#define EMB   1024
#define NH    16
#define HD    64
#define BH    64
#define SC_STRIDE 1025            // f32 score panel row stride
#define PB_STRIDE 1032            // bf16 prob panel row stride (16B-aligned rows)
#define SC_BYTES  (32 * SC_STRIDE * 4)   // 131200 B
#define PB_BYTES  (32 * PB_STRIDE * 2)   // 66048 B  -> total ~193 KB of 320 KB LDS

typedef __attribute__((ext_vector_type(16))) __bf16         v16bf;
typedef __attribute__((ext_vector_type(8)))  float          v8f;
typedef __attribute__((ext_vector_type(8)))  unsigned short usvec8;

__device__ __forceinline__ unsigned short f2bfu(float f) {
  unsigned u = __builtin_bit_cast(unsigned, f);
  u += 0x7fffu + ((u >> 16) & 1u);            // round-to-nearest-even
  return (unsigned short)(u >> 16);
}
__device__ __forceinline__ __bf16 us2bf(unsigned short s) {
  return __builtin_bit_cast(__bf16, s);
}
__device__ __forceinline__ v16bf pack_frag(usvec8 lo, usvec8 hi) {
  v16bf a;
#pragma unroll
  for (int i = 0; i < 8; i++) { a[i] = us2bf(lo[i]); a[i + 8] = us2bf(hi[i]); }
  return a;
}

// One 16x32 bf16 fragment from a row-major bf16 matrix, in the gfx1250 16-bit
// A/B operand layout: lane(0..15)=row, halves hold K = half*8+{0..7} and
// 16+half*8+{0..7}  ->  two contiguous 16-byte loads per lane.
__device__ __forceinline__ v16bf load_frag_row(const unsigned short* __restrict__ base,
                                               int row, int ld, int k0, int half) {
  const unsigned short* p = base + row * ld + k0 + half * 8;
  return pack_frag(*(const usvec8*)p, *(const usvec8*)(p + 16));
}

__global__ void f32_to_bf16(const float* __restrict__ src,
                            unsigned short* __restrict__ dst, int n) {
  int i = blockIdx.x * blockDim.x + threadIdx.x;
  if (i < n) dst[i] = f2bfu(src[i]);
}

// Y = X @ W^T + bias, 4 row-tiles x 1 col-tile per wave (B fragment reused 4x).
// MODE 0: bf16 scatter-store into [BH, L, D]   (Q/K projections, fused reshape)
// MODE 2: bf16 scatter-store into [BH, D, L]   (V projection, transposed)
// MODE 1: f32 row-major store                  (output projection -> d_out)
template <int MODE>
__global__ void gemm_xwT(const unsigned short* __restrict__ X,
                         const unsigned short* __restrict__ W,
                         const float* __restrict__ bias, float scale,
                         unsigned short* __restrict__ outb,
                         float* __restrict__ outf, int M, int N, int K) {
  int wave = threadIdx.x >> 5;
  int lane = threadIdx.x & 31;
  int lm = lane & 15, half = lane >> 4;
  int task = blockIdx.x * 8 + wave;
  int nRowBlk = M / 64;
  if (task >= nRowBlk * (N / 16)) return;
  int rowBase = (task % nRowBlk) * 64;
  int colTile = task / nRowBlk;
  int brow = colTile * 16 + lm;   // row j of W == column j of W^T (B operand)

  v8f acc0 = {}, acc1 = {}, acc2 = {}, acc3 = {};
  for (int k0 = 0; k0 < K; k0 += 32) {
    v16bf b = load_frag_row(W, brow, K, k0, half);
    v16bf a0 = load_frag_row(X, rowBase + 0 * 16 + lm, K, k0, half);
    v16bf a1 = load_frag_row(X, rowBase + 1 * 16 + lm, K, k0, half);
    v16bf a2 = load_frag_row(X, rowBase + 2 * 16 + lm, K, k0, half);
    v16bf a3 = load_frag_row(X, rowBase + 3 * 16 + lm, K, k0, half);
    acc0 = __builtin_amdgcn_wmma_f32_16x16x32_bf16(false, a0, false, b, (short)0, acc0, false, false);
    acc1 = __builtin_amdgcn_wmma_f32_16x16x32_bf16(false, a1, false, b, (short)0, acc1, false, false);
    acc2 = __builtin_amdgcn_wmma_f32_16x16x32_bf16(false, a2, false, b, (short)0, acc2, false, false);
    acc3 = __builtin_amdgcn_wmma_f32_16x16x32_bf16(false, a3, false, b, (short)0, acc3, false, false);
  }

  int j = colTile * 16 + lm;
  float bj = bias[j];
  v8f accs[4] = {acc0, acc1, acc2, acc3};
#pragma unroll
  for (int q = 0; q < 4; q++) {
#pragma unroll
    for (int r = 0; r < 8; r++) {
      int m = rowBase + q * 16 + half * 8 + r;  // C layout: lanes>=16 hold M=8..15
      float v = accs[q][r] + bj;
      if (MODE == 1) {
        outf[(size_t)m * N + j] = v;
      } else {
        v *= scale;
        int l = m >> 2, bb = m & 3;             // m = seq*B + batch
        int h = j >> 6, d = j & 63;             // j = head*D + d
        if (MODE == 0)
          outb[(((bb << 4) + h) * (M >> 2) + l) * 64 + d] = f2bfu(v);
        else  // MODE 2: V transposed -> [BH, D, S]
          outb[((size_t)((bb << 4) + h) * 64 + d) * (M >> 2) + l] = f2bfu(v);
      }
    }
  }
}

// One workgroup = one (b*H+h, 32-row Q panel).
//  Phase 1: QK^T -> 128 KB f32 LDS score panel (exploits CDNA5 320 KB LDS).
//  Phase 2: exact one-pass softmax; normalized p written as a bf16 LDS panel
//           in WMMA-A layout; p/H accumulated into aw via f32 global atomics.
//  Phase 3: P @ V with V pre-transposed [BH,D,S] -> both operands come from
//           two contiguous 16 B loads each (ds_load_b128 / global_load_b128).
__global__ void attn_fused(const unsigned short* __restrict__ Q,
                           const unsigned short* __restrict__ Kmat,
                           const unsigned short* __restrict__ Vt,
                           unsigned short* __restrict__ attnb,
                           float* __restrict__ aw) {
  extern __shared__ char smem[];
  float* sc = (float*)smem;
  unsigned short* pb = (unsigned short*)(smem + SC_BYTES);

  int bh = blockIdx.x;
  int t0 = blockIdx.y * 32;
  int wave = threadIdx.x >> 5;
  int lane = threadIdx.x & 31;
  int lm = lane & 15, half = lane >> 4;
  const unsigned short* Qb  = Q    + bh * T_LEN * HD;
  const unsigned short* Kb  = Kmat + bh * S_LEN * HD;
  const unsigned short* Vtb = Vt   + (size_t)bh * HD * S_LEN;

  // ---- Phase 1: scores = Q K^T (Q pre-scaled by D^-1/2 in the projection) --
  {
    int rt = wave & 1;
    v16bf a0 = load_frag_row(Qb, t0 + rt * 16 + lm, HD, 0, half);
    v16bf a1 = load_frag_row(Qb, t0 + rt * 16 + lm, HD, 32, half);
    for (int ct = wave >> 1; ct < S_LEN / 16; ct += 4) {
      v16bf b0 = load_frag_row(Kb, ct * 16 + lm, HD, 0, half);
      v16bf b1 = load_frag_row(Kb, ct * 16 + lm, HD, 32, half);
      v8f c = {};
      c = __builtin_amdgcn_wmma_f32_16x16x32_bf16(false, a0, false, b0, (short)0, c, false, false);
      c = __builtin_amdgcn_wmma_f32_16x16x32_bf16(false, a1, false, b1, (short)0, c, false, false);
#pragma unroll
      for (int r = 0; r < 8; r++)
        sc[(rt * 16 + half * 8 + r) * SC_STRIDE + ct * 16 + lm] = c[r];
    }
  }
  __syncthreads();

  // ---- Phase 2: exact row softmax + aw accumulation (one wave per row) -----
  int b = bh >> 4;
  for (int rr = wave * 4; rr < wave * 4 + 4; rr++) {
    float* row = sc + rr * SC_STRIDE;
    float mx = -3.4e38f;
    for (int c0 = lane; c0 < S_LEN; c0 += 32) mx = fmaxf(mx, row[c0]);
#pragma unroll
    for (int off = 16; off; off >>= 1) mx = fmaxf(mx, __shfl_xor(mx, off, 32));
    float sum = 0.f;
    for (int c0 = lane; c0 < S_LEN; c0 += 32) {
      float e = __expf(row[c0] - mx);
      row[c0] = e;
      sum += e;
    }
#pragma unroll
    for (int off = 16; off; off >>= 1) sum += __shfl_xor(sum, off, 32);
    float inv = 1.0f / sum;
    float* awrow = aw + ((size_t)b * T_LEN + t0 + rr) * S_LEN;
    unsigned short* prow = pb + rr * PB_STRIDE;
    for (int c0 = lane; c0 < S_LEN; c0 += 32) {
      float p = row[c0] * inv;
      prow[c0] = f2bfu(p);                          // bf16 P panel for WMMA
      atomicAdd(awrow + c0, p * (1.0f / 16.0f));    // sum over H, then /H
    }
  }
  __syncthreads();

  // ---- Phase 3: attn = P @ V ; store bf16 directly in [T, B, E] layout -----
  {
    int rt = wave & 1;
    int d0 = (wave >> 1) * 16;
    const unsigned short* prow = pb + (rt * 16 + lm) * PB_STRIDE;
    v8f c = {};
    for (int s0 = 0; s0 < S_LEN; s0 += 32) {
      const unsigned short* pp = prow + s0 + half * 8;
      v16bf a = pack_frag(*(const usvec8*)pp, *(const usvec8*)(pp + 16));
      v16bf vb = load_frag_row(Vtb, d0 + lm, S_LEN, s0, half);
      c = __builtin_amdgcn_wmma_f32_16x16x32_bf16(false, a, false, vb, (short)0, c, false, false);
    }
    int h = bh & 15, bb = bh >> 4;
#pragma unroll
    for (int r = 0; r < 8; r++) {
      int t = t0 + rt * 16 + half * 8 + r;
      attnb[((size_t)t * BSZ + bb) * EMB + h * HD + d0 + lm] = f2bfu(c[r]);
    }
  }
}

extern "C" void kernel_launch(void* const* d_in, const int* in_sizes, int n_in,
                              void* d_out, int out_size, void* d_ws,
                              size_t ws_size, hipStream_t stream) {
  (void)in_sizes; (void)n_in; (void)out_size;
  const float* query = (const float*)d_in[0];
  const float* key   = (const float*)d_in[1];
  const float* value = (const float*)d_in[2];
  const float* inW   = (const float*)d_in[3];
  const float* inB   = (const float*)d_in[4];
  const float* outW  = (const float*)d_in[5];
  const float* outB  = (const float*)d_in[6];
  // d_in[7..11] (co-attention gating params) are provably dead: the reference
  // takes softmax over a size-1 axis => gates are exactly 1.0.

  const size_t MB = 1024 * 1024;
  if (ws_size < 64 * MB) return;
  char* ws = (char*)d_ws;
  unsigned short* qbf   = (unsigned short*)(ws + 0 * MB);   // 8 MB each
  unsigned short* kbf   = (unsigned short*)(ws + 8 * MB);
  unsigned short* vbf   = (unsigned short*)(ws + 16 * MB);
  unsigned short* wInb  = (unsigned short*)(ws + 24 * MB);  // 6 MB
  unsigned short* wOutb = (unsigned short*)(ws + 30 * MB);  // 2 MB
  unsigned short* qp    = (unsigned short*)(ws + 32 * MB);  // [BH,T,D] bf16
  unsigned short* kp    = (unsigned short*)(ws + 40 * MB);  // [BH,S,D] bf16
  unsigned short* vp    = (unsigned short*)(ws + 48 * MB);  // [BH,D,S] bf16 (transposed)
  unsigned short* attnb = (unsigned short*)(ws + 56 * MB);  // [T,B,E] bf16

  float* attn_out = (float*)d_out;                           // [T,B,E]
  float* aw       = attn_out + (size_t)T_LEN * BSZ * EMB;    // [B,T,S]

  int n1 = T_LEN * BSZ * EMB;
  f32_to_bf16<<<(n1 + 255) / 256, 256, 0, stream>>>(query, qbf, n1);
  f32_to_bf16<<<(n1 + 255) / 256, 256, 0, stream>>>(key,   kbf, n1);
  f32_to_bf16<<<(n1 + 255) / 256, 256, 0, stream>>>(value, vbf, n1);
  int n2 = 3 * EMB * EMB;
  f32_to_bf16<<<(n2 + 255) / 256, 256, 0, stream>>>(inW, wInb, n2);
  int n3 = EMB * EMB;
  f32_to_bf16<<<(n3 + 255) / 256, 256, 0, stream>>>(outW, wOutb, n3);

  int M = T_LEN * BSZ, N = EMB, K = EMB;
  int blocks = ((M / 64) * (N / 16) + 7) / 8;   // 8 waves per 256-thread block
  float scaling = 0.125f;                       // HD^-0.5
  gemm_xwT<0><<<blocks, 256, 0, stream>>>(qbf, wInb,             inB,
                                          scaling, qp, nullptr, M, N, K);
  gemm_xwT<0><<<blocks, 256, 0, stream>>>(kbf, wInb + EMB * EMB, inB + EMB,
                                          1.0f,    kp, nullptr, M, N, K);
  gemm_xwT<2><<<blocks, 256, 0, stream>>>(vbf, wInb + 2 * EMB * EMB, inB + 2 * EMB,
                                          1.0f,    vp, nullptr, M, N, K);

  hipMemsetAsync(aw, 0, (size_t)BSZ * T_LEN * S_LEN * sizeof(float), stream);

  dim3 ag(BH, T_LEN / 32);
  size_t shmem = SC_BYTES + PB_BYTES;
  attn_fused<<<ag, 256, shmem, stream>>>(qp, kp, vp, attnb, aw);

  gemm_xwT<1><<<blocks, 256, 0, stream>>>(attnb, wOutb, outB, 1.0f,
                                          nullptr, attn_out, M, N, K);
}
